// GraphAttentionLayer_5403068858688
// MI455X (gfx1250) — matrix operand order repaired and checked by reading, not compiled, not run on hardware
//
#include <hip/hip_runtime.h>
#include <hip/hip_bf16.h>
#include <cmath>

typedef __attribute__((ext_vector_type(16))) _Float16 v16h;
typedef __attribute__((ext_vector_type(8)))  float    v8f;
typedef __attribute__((ext_vector_type(4)))  unsigned u32x4;
typedef __attribute__((ext_vector_type(8)))  unsigned u32x8;

#define GAT_ALPHA 0.2f
constexpr int Bb = 8;
constexpr int Nn = 2048;
constexpr int Ff = 256;                 // F_IN == F_OUT == 256
constexpr unsigned LDS_ROW   = 528;     // 512B data + 16B TDM pad -> conflict-free ds_load_b128
constexpr unsigned LDS_TILE  = 32 * LDS_ROW;   // 16,896 B per buffer

// ---------------------------------------------------------------------------
// K0: transpose W (f32 [O,F]) -> Wt (f16 [F,O]) so WMMA B-tiles are contiguous
// ---------------------------------------------------------------------------
__global__ __launch_bounds__(256) void pack_w(const float* __restrict__ W,
                                              _Float16* __restrict__ Wt) {
  int idx = blockIdx.x * 256 + threadIdx.x;   // 65536 elements
  int o = idx / Ff, f = idx % Ff;             // coalesced read of W
  Wt[f * Ff + o] = (_Float16)W[idx];
}

// ---------------------------------------------------------------------------
// K0b: pack adj (int32, 134MB) into bitmask (4.2MB) with wave32 ballot.
// One HBM pass over adj; both softmax passes then read only the bitmask.
// ---------------------------------------------------------------------------
__global__ __launch_bounds__(256) void pack_adj(const int* __restrict__ adj,
                                                unsigned* __restrict__ mask) {
  long long w = (long long)blockIdx.x * 8 + (threadIdx.x >> 5); // word index
  int lane = threadIdx.x & 31;
  int v = adj[w * 32 + lane];
#if __has_builtin(__builtin_amdgcn_ballot_w32)
  unsigned bal = __builtin_amdgcn_ballot_w32(v != 0);
#else
  unsigned bal = (unsigned)__ballot(v != 0);
#endif
  if (lane == 0) mask[w] = bal;
}

// ---------------------------------------------------------------------------
// K1: Wh = h @ W^T  via v_wmma_f32_16x16x32_f16.
// Each wave: 16 rows x 64 cols (4 C tiles). Stores f32 + f16 copies.
// ---------------------------------------------------------------------------
__global__ __launch_bounds__(256) void wh_gemm(const float* __restrict__ h,
                                               const _Float16* __restrict__ Wt,
                                               float* __restrict__ Wh,
                                               _Float16* __restrict__ Whh) {
  int wave = blockIdx.x * 8 + (threadIdx.x >> 5);
  int lane = threadIdx.x & 31;
  int tile = wave >> 2;                 // 0..1023 : 16-row tiles of B*N
  int oq   = (wave & 3) * 64;           // o-quarter base
  int row0 = tile * 16;
  int m    = lane & 15;
  int khalf = (lane >> 4) * 8;          // A-layout: lane half selects K offset

  v8f acc[4] = {};
  const float* hrow = h + (long long)(row0 + m) * Ff;
  for (int k0 = 0; k0 < Ff; k0 += 32) {
    v16h a;
    const float* p1 = hrow + k0 + khalf;        // K = k0+khalf .. +7
    const float* p2 = p1 + 16;                  // K = k0+khalf+16 .. +7
#pragma unroll
    for (int t = 0; t < 8; ++t) { a[t] = (_Float16)p1[t]; a[8 + t] = (_Float16)p2[t]; }
#pragma unroll
    for (int ot = 0; ot < 4; ++ot) {
      // B layout (f16, K=32): lane%16 = N col, elements = K (same order as A)
      v16h b;
      const _Float16* wb = Wt + (long long)(k0 + khalf) * Ff + oq + ot * 16 + m;
#pragma unroll
      for (int t = 0; t < 8; ++t) { b[t] = wb[t * Ff]; b[8 + t] = wb[(16 + t) * Ff]; }
      acc[ot] = __builtin_amdgcn_wmma_f32_16x16x32_f16(false, a, false, b,
                                                       (short)0, acc[ot], false, false);
    }
  }
  // C layout: VGPR r -> row (r + 8*(lane>=16)), col = lane%16
  int rbase = row0 + ((lane >> 4) ? 8 : 0);
#pragma unroll
  for (int ot = 0; ot < 4; ++ot) {
    int col = oq + ot * 16 + m;
#pragma unroll
    for (int r = 0; r < 8; ++r) {
      float v = acc[ot][r];
      long long off = (long long)(rbase + r) * Ff + col;
      Wh[off]  = v;
      Whh[off] = (_Float16)v;
    }
  }
}

// ---------------------------------------------------------------------------
// K2: s1 = Wh . a1, s2 = Wh . a2 per row; wave32 shfl_xor reduction
// ---------------------------------------------------------------------------
__global__ __launch_bounds__(256) void s_kernel(const float* __restrict__ Wh,
                                                const float* __restrict__ a,
                                                float* __restrict__ s1,
                                                float* __restrict__ s2) {
  long long row = (long long)blockIdx.x * 8 + (threadIdx.x >> 5);
  int lane = threadIdx.x & 31;
  const float* r = Wh + row * Ff;
  float d1 = 0.f, d2 = 0.f;
#pragma unroll
  for (int c = 0; c < Ff; c += 32) {
    float v = r[c + lane];
    d1 += v * a[c + lane];
    d2 += v * a[Ff + c + lane];
  }
#pragma unroll
  for (int off = 16; off; off >>= 1) {
    d1 += __shfl_xor(d1, off, 32);
    d2 += __shfl_xor(d2, off, 32);
  }
  if (lane == 0) { s1[row] = d1; s2[row] = d2; }
}

// ---------------------------------------------------------------------------
// K3: pass 1 — per-row online softmax stats (max, sum) over masked scores.
// Lanes L and L+16 hold the same row (A layout); combined with xor-16 shuffle.
// ---------------------------------------------------------------------------
__global__ __launch_bounds__(256) void stats_kernel(const float* __restrict__ s1,
                                                    const float* __restrict__ s2,
                                                    const unsigned* __restrict__ maskbits,
                                                    float* __restrict__ Mrow,
                                                    float* __restrict__ Lrow) {
  int wave = blockIdx.x * 8 + (threadIdx.x >> 5);  // 0..1023
  int lane = threadIdx.x & 31;
  int b  = wave >> 7;
  int i0 = (wave & 127) * 16;
  int i  = i0 + (lane & 15);
  int off1 = (lane >> 4) * 8;

  float s1i = s1[(long long)b * Nn + i];
  const float* s2b = s2 + (long long)b * Nn;
  const unsigned* mrow = maskbits + ((long long)b * Nn + i) * (Nn / 32);

  float m = -__builtin_inff();
  float l = 0.f;
  for (int j0 = 0; j0 < Nn; j0 += 32) {
    unsigned mw = mrow[j0 >> 5];
#pragma unroll
    for (int g = 0; g < 2; ++g) {
#pragma unroll
      for (int t = 0; t < 8; ++t) {
        int kk = off1 + g * 16 + t;
        if ((mw >> kk) & 1u) {
          float s = s1i + s2b[j0 + kk];
          float e = s > 0.f ? s : GAT_ALPHA * s;
          if (e > m) { l = l * __expf(m - e) + 1.f; m = e; }
          else       { l += __expf(e - m); }
        }
      }
    }
  }
  float m2 = __shfl_xor(m, 16, 32);
  float l2 = __shfl_xor(l, 16, 32);
  float mm = fmaxf(m, m2);
  float ll = ((m  == -__builtin_inff()) ? 0.f : l  * __expf(m  - mm)) +
             ((m2 == -__builtin_inff()) ? 0.f : l2 * __expf(m2 - mm));
  if (lane < 16) {
    Mrow[(long long)b * Nn + i] = mm;
    Lrow[(long long)b * Nn + i] = ll;
  }
}

// ---------------------------------------------------------------------------
// K4: pass 2 — out = att @ Wh.
// 512-thread workgroup = 16 waves = 8 i-tiles x 2 o-halves sharing one
// TDM-staged 32x256 Wh tile per j-chunk (8x less L2 traffic). Double-buffered:
// wave 0 issues tensor_load_to_lds for chunk c+1, all waves compute chunk c
// from LDS (bank-conflict-free via TDM hardware padding), s_wait_tensorcnt,
// barrier, repeat.
// ---------------------------------------------------------------------------
__global__ __launch_bounds__(512) void attn_out(const float* __restrict__ s1,
                                                const float* __restrict__ s2,
                                                const unsigned* __restrict__ maskbits,
                                                const float* __restrict__ Mrow,
                                                const float* __restrict__ Lrow,
                                                const _Float16* __restrict__ Whh,
                                                float* __restrict__ out) {
  __shared__ __align__(16) char lbuf[2][LDS_TILE];

  int tid  = threadIdx.x;
  int lane = tid & 31;
  int w    = tid >> 5;                   // wave in block: 0..15
  int ohalf = (w & 1) * 128;
  int tile  = blockIdx.x * 8 + (w >> 1); // 0..1023
  int b  = tile >> 7;                    // whole block shares one batch b
  int i0 = (tile & 127) * 16;
  int i  = i0 + (lane & 15);
  int off1 = (lane >> 4) * 8;

  float s1i = s1[(long long)b * Nn + i];
  float mi  = Mrow[(long long)b * Nn + i];
  float li  = Lrow[(long long)b * Nn + i];
  float inv = li > 0.f ? 1.f / li : 0.f;
  const float* s2b = s2 + (long long)b * Nn;
  const unsigned* mrow = maskbits + ((long long)b * Nn + i) * (Nn / 32);
  const _Float16* WhB = Whh + (long long)b * Nn * Ff;

  unsigned ldsBase = (unsigned)(uintptr_t)&lbuf[0][0];

  // Issue one TDM load of Wh[b, j0 .. j0+31, 0..255] (f16) into LDS buffer.
  // D# per CDNA5 ISA ch.8: data_size=8B units; tile 64x32; row stride 64 units;
  // pad_enable, pad every 128 DWORDs (512B) by 4 DWORDs (16B) -> 528B LDS rows.
  auto tdm_issue = [&](int j0, int bufIdx) {
    unsigned long long ga =
        (unsigned long long)(uintptr_t)(WhB + (long long)j0 * Ff);
    u32x4 g0;
    g0[0] = 1u;                                          // count=1, user mode
    g0[1] = ldsBase + (unsigned)bufIdx * LDS_TILE;       // lds_addr
    g0[2] = (unsigned)ga;                                // global_addr[31:0]
    g0[3] = ((unsigned)(ga >> 32) & 0x01FFFFFFu) | 0x80000000u; // addr[56:32] | type=2
    u32x8 g1;
    g1[0] = (3u << 16)     // data_size = 8B
          | (1u << 20)     // pad_enable
          | (6u << 22)     // pad_interval: 512B
          | (3u << 25);    // pad_amount: 4 DWORDs (16B)
    g1[1] = 64u << 16;     // tensor_dim0 = 64 units (one 512B row)
    g1[2] = 2048u << 16;   // tensor_dim1 = N rows
    g1[3] = 64u << 16;     // tile_dim0 = 64 units
    g1[4] = 32u;           // tile_dim1 = 32 rows
    g1[5] = 64u;           // tensor_dim0_stride = 64 units (contiguous rows)
    g1[6] = 0u;
    g1[7] = 0u;
    asm volatile("tensor_load_to_lds %0, %1" :: "s"(g0), "s"(g1) : "memory");
  };

  v8f acc[8] = {};

  // Prologue: stage chunk 0.
  if (tid < 32) {
    tdm_issue(0, 0);
#if __has_builtin(__builtin_amdgcn_s_wait_tensorcnt)
    __builtin_amdgcn_s_wait_tensorcnt(0);
#else
    asm volatile("s_wait_tensorcnt 0x0" ::: "memory");
#endif
  }
  __syncthreads();

  for (int chunk = 0; chunk < Nn / 32; ++chunk) {
    int cur = chunk & 1;
    int j0  = chunk * 32;
    if (tid < 32 && chunk + 1 < Nn / 32)
      tdm_issue(j0 + 32, cur ^ 1);       // overlap next tile with compute

    // Build normalized attention probabilities directly in A-fragment layout.
    v16h a;
#pragma unroll
    for (int g = 0; g < 2; ++g) {
      const float* p = s2b + j0 + off1 + g * 16;
      float4 x0 = *(const float4*)p;
      float4 x1 = *(const float4*)(p + 4);
      float vs[8] = { x0.x, x0.y, x0.z, x0.w, x1.x, x1.y, x1.z, x1.w };
      unsigned mw = mrow[chunk];
#pragma unroll
      for (int t = 0; t < 8; ++t) {
        int kk = off1 + g * 16 + t;
        float pr = 0.f;
        if ((mw >> kk) & 1u) {
          float s = s1i + vs[t];
          float e = s > 0.f ? s : GAT_ALPHA * s;
          pr = __expf(e - mi) * inv;
        }
        a[g * 8 + t] = (_Float16)pr;
      }
    }

    // B fragments from LDS: lane%16 = N col, elements = K rows (same order as A).
    const char* bbase = &lbuf[cur][0] + (unsigned)(ohalf + (lane & 15)) * 2;
#pragma unroll
    for (int ot = 0; ot < 8; ++ot) {
      union { uint4 q[2]; v16h v; } f;
#pragma unroll
      for (int t = 0; t < 8; ++t) {
        f.v[t]     = *(const _Float16*)(bbase + (unsigned)(off1 + t) * LDS_ROW + ot * 32);
        f.v[8 + t] = *(const _Float16*)(bbase + (unsigned)(off1 + 16 + t) * LDS_ROW + ot * 32);
      }
      acc[ot] = __builtin_amdgcn_wmma_f32_16x16x32_f16(false, a, false, f.v,
                                                       (short)0, acc[ot], false, false);
    }

    if (tid < 32) {
#if __has_builtin(__builtin_amdgcn_s_wait_tensorcnt)
      __builtin_amdgcn_s_wait_tensorcnt(0);
#else
      asm volatile("s_wait_tensorcnt 0x0" ::: "memory");
#endif
    }
    __syncthreads();
  }

  int rbase = i0 + ((lane >> 4) ? 8 : 0);
  float* orow = out + (long long)b * Nn * Ff;
#pragma unroll
  for (int ot = 0; ot < 8; ++ot) {
    int col = ohalf + ot * 16 + (lane & 15);
#pragma unroll
    for (int r = 0; r < 8; ++r)
      orow[(long long)(rbase + r) * Ff + col] = acc[ot][r];
  }
}

// ---------------------------------------------------------------------------
// Host launcher
// ---------------------------------------------------------------------------
extern "C" void kernel_launch(void* const* d_in, const int* in_sizes, int n_in,
                              void* d_out, int out_size, void* d_ws, size_t ws_size,
                              hipStream_t stream) {
  const float* h   = (const float*)d_in[0];   // [B,N,F_IN] f32
  const int*   adj = (const int*)d_in[1];     // [B,N,N] i32
  const float* W   = (const float*)d_in[2];   // [F_OUT,F_IN] f32
  const float* a   = (const float*)d_in[3];   // [1, 2*F_OUT] f32
  float* out = (float*)d_out;

  // Workspace layout (bytes)
  char* ws = (char*)d_ws;
  float*     Wh_f32 = (float*)(ws);                                 // 16,777,216
  _Float16*  Wh_f16 = (_Float16*)(ws + 16777216);                   //  8,388,608
  _Float16*  Wt_f16 = (_Float16*)(ws + 25165824);                   //    131,072
  float*     s1     = (float*)(ws + 25296896);                      //     65,536
  float*     s2     = (float*)(ws + 25362432);                      //     65,536
  float*     Mrow   = (float*)(ws + 25427968);                      //     65,536
  float*     Lrow   = (float*)(ws + 25493504);                      //     65,536
  unsigned*  mbits  = (unsigned*)(ws + 25559040);                   //  4,194,304

  pack_w   <<<256,    256, 0, stream>>>(W, Wt_f16);
  pack_adj <<<131072, 256, 0, stream>>>(adj, mbits);                // 1,048,576 words
  wh_gemm  <<<512,    256, 0, stream>>>(h, Wt_f16, Wh_f32, Wh_f16); // 4096 waves
  s_kernel <<<2048,   256, 0, stream>>>(Wh_f32, a, s1, s2);         // 16384 waves
  stats_kernel<<<128, 256, 0, stream>>>(s1, s2, mbits, Mrow, Lrow); // 1024 waves
  attn_out <<<128,    512, 0, stream>>>(s1, s2, mbits, Mrow, Lrow, Wh_f16, out); // 128 blocks x 16 waves
}
